// Net_68195490725951
// MI455X (gfx1250) — matrix operand order repaired and checked by reading, not compiled, not run on hardware
//
#include <hip/hip_runtime.h>
#include <hip/hip_bf16.h>
#include <cstdint>

// ---------------------------------------------------------------------------
// MI455X (gfx1250) iPASSR-style stereo SR network.
//  * Activations: bf16 NHWC -> im2col implicit in WMMA fragment addressing.
//  * All convs + window attention: V_WMMA_F32_16X16X32_BF16, fp32 accumulate.
//  * Input rows  -> LDS via GLOBAL_LOAD_ASYNC_TO_LDS_B64   (ASYNCcnt path)
//  * Weight tile -> LDS via TENSOR_LOAD_TO_LDS (TDM 2D D#) (TENSORcnt path)
//  * Working set is L2-resident (192MB @ 23.3TB/s HBM) -> matrix-bound.
// ---------------------------------------------------------------------------

typedef __attribute__((ext_vector_type(16))) __bf16 bf16x16;
typedef __attribute__((ext_vector_type(8)))  float  v8f;
typedef __attribute__((ext_vector_type(4)))  unsigned uint4v;
typedef __attribute__((ext_vector_type(4)))  int     int4v;
typedef __attribute__((ext_vector_type(8)))  int     int8v;

union FragU { unsigned u[8]; bf16x16 v; };

#if __has_builtin(__builtin_amdgcn_tensor_load_to_lds)
#define HAVE_TDM 1
#else
#define HAVE_TDM 0
#endif

__device__ inline unsigned short f2bf(float f) {
  union { float f; unsigned u; } x; x.f = f;
  unsigned r = x.u + 0x7fffu + ((x.u >> 16) & 1u);
  if ((x.u & 0x7f800000u) == 0x7f800000u) r = x.u;
  return (unsigned short)(r >> 16);
}
__device__ inline float bf2f(unsigned short u) {
  union { unsigned u; float f; } x; x.u = ((unsigned)u) << 16; return x.f;
}

__device__ inline v8f vzero8() {
  v8f z;
#pragma unroll
  for (int i = 0; i < 8; ++i) z[i] = 0.f;
  return z;
}

__device__ inline v8f wmma_bf16(bf16x16 a, bf16x16 b, v8f c) {
  return __builtin_amdgcn_wmma_f32_16x16x32_bf16(false, a, false, b, (short)0, c,
                                                 false, false);
}

// async global->LDS copy of 8 bytes (per-lane DMA, ASYNCcnt tracked).
// Low 32 bits of a generic shared pointer are the LDS byte offset (ISA 10.2).
__device__ inline void async_b64(const void* gsrc, const void* lds_ptr) {
  unsigned loff = (unsigned)(uintptr_t)lds_ptr;
  asm volatile("global_load_async_to_lds_b64 %0, %1, off"
               :: "v"(loff), "v"(gsrc) : "memory");
}
__device__ inline void wait_async0() {
#if __has_builtin(__builtin_amdgcn_s_wait_asynccnt)
  __builtin_amdgcn_s_wait_asynccnt(0);
#else
  asm volatile("s_wait_asynccnt 0x0" ::: "memory");
#endif
}
__device__ inline void wait_tensor0() {
#if __has_builtin(__builtin_amdgcn_s_wait_tensorcnt)
  __builtin_amdgcn_s_wait_tensorcnt(0);
#else
  asm volatile("s_wait_tensorcnt 0x0" ::: "memory");
#endif
}

#if HAVE_TDM
// TDM: DMA a 2D tile (rows x rowLen bf16, row stride rowStride elements) from
// global into LDS. rowsAvail bounds OOB rows -> TDM zero-fills them.
// D# bitfields per CDNA5 ISA ch.8 (group0: count/lds/global/type; group1:
// data_size, tensor_dim0/1, tile_dim0/1, tensor_dim0_stride).
__device__ inline void tdm_load_2d(const void* gsrc, void* lds_dst, int rowLen,
                                   int rows, int rowsAvail, int rowStride) {
  uint64_t ga = (uint64_t)(uintptr_t)gsrc;
  unsigned lo = (unsigned)(uintptr_t)lds_dst;
  uint4v g0;
  g0.x = 1u;                                       // count=1, user mode
  g0.y = lo;                                       // lds_addr
  g0.z = (unsigned)ga;                             // global_addr[31:0]
  g0.w = ((unsigned)(ga >> 32) & 0x01ffffffu) | (2u << 30);  // addr[56:32]|type=2
  int8v g1;
  g1[0] = 0x00010000;                              // data_size=1 (2 bytes)
  g1[1] = (int)(((unsigned)rowLen & 0xffffu) << 16);           // tensor_dim0 lo
  g1[2] = (int)((((unsigned)rowLen >> 16) & 0xffffu) |
                (((unsigned)rowsAvail & 0xffffu) << 16));      // dim0 hi|dim1 lo
  g1[3] = (int)((((unsigned)rowsAvail >> 16) & 0xffffu) |
                (((unsigned)rowLen & 0xffffu) << 16));         // dim1 hi|tile0
  g1[4] = rows & 0xffff;                           // tile_dim1 (tile_dim2=0)
  g1[5] = rowStride;                               // tensor_dim0_stride lo32
  g1[6] = 0;
  g1[7] = 0;
  int4v z4 = {0, 0, 0, 0};
#if defined(__clang_major__) && __clang_major__ >= 23
  int8v z8 = {0, 0, 0, 0, 0, 0, 0, 0};
  __builtin_amdgcn_tensor_load_to_lds(g0, g1, z4, z4, z8, 0);
#else
  __builtin_amdgcn_tensor_load_to_lds(g0, g1, z4, z4, 0);
#endif
}
#endif

// A fragment (16x32 bf16 MxK) from [row][K] bf16, K-pairs contiguous.
__device__ inline bf16x16 fragA_mem(const unsigned short* rowPtr, int kb, int lane,
                                    bool rowOk) {
  FragU f;
  int lh = lane >> 4;
#pragma unroll
  for (int v = 0; v < 8; ++v) {
    int k0 = kb + 2 * v + 16 * lh;
    f.u[v] = rowOk ? *(const unsigned*)(rowPtr + k0) : 0u;
  }
  return f.v;
}
// B fragment (32x16 bf16 KxN) from [col][K] bf16 (B-transposed row-major).
__device__ inline bf16x16 fragB_mem(const unsigned short* colPtr, int kb, int lane) {
  FragU f;
  int lh = lane >> 4;
#pragma unroll
  for (int v = 0; v < 8; ++v)
    f.u[v] = *(const unsigned*)(colPtr + kb + 2 * v + 16 * lh);
  return f.v;
}

// ------------------------------ small kernels ------------------------------

__global__ void k_nchw2nhwc(const float* __restrict__ in,
                            unsigned short* __restrict__ out, int C, int HW) {
  int i = blockIdx.x * blockDim.x + threadIdx.x;
  if (i >= C * HW) return;
  int p = i / C, c = i - p * C;
  out[i] = f2bf(in[(size_t)c * HW + p]);
}

__global__ void k_catcpy(const unsigned short* __restrict__ s, int sS,
                         unsigned short* __restrict__ d, int dS, int dOff,
                         int C, int HW) {
  int i = blockIdx.x * blockDim.x + threadIdx.x;
  if (i >= HW * C) return;
  int p = i / C, c = i - p * C;
  d[(size_t)p * dS + dOff + c] = s[(size_t)p * sS + c];
}

__global__ void k_affine_bf(const unsigned short* __restrict__ x,
                            const float* __restrict__ g, const float* __restrict__ b,
                            unsigned short* __restrict__ y, int C, int total) {
  int i = blockIdx.x * blockDim.x + threadIdx.x;
  if (i >= total) return;
  int c = i % C;
  y[i] = f2bf(bf2f(x[i]) * (g[c] * rsqrtf(1.0f + 1e-5f)) + b[c]);
}

__global__ void k_gap_bf(const unsigned short* __restrict__ x, float* __restrict__ y,
                         int C, int HW) {
  __shared__ float red[256];
  int c = blockIdx.x;
  float s = 0.f;
  for (int p = threadIdx.x; p < HW; p += 256) s += bf2f(x[(size_t)p * C + c]);
  red[threadIdx.x] = s;
  __syncthreads();
  for (int st = 128; st > 0; st >>= 1) {
    if ((int)threadIdx.x < st) red[threadIdx.x] += red[threadIdx.x + st];
    __syncthreads();
  }
  if (threadIdx.x == 0) y[c] = red[0] / (float)HW;
}

__global__ void k_ca_mlp(const float* __restrict__ g, const float* __restrict__ w1,
                         const float* __restrict__ b1, const float* __restrict__ w2,
                         const float* __restrict__ b2, float* __restrict__ sc) {
  __shared__ float gv[128];
  __shared__ float hid[8];
  int t = threadIdx.x;
  gv[t] = g[t];
  __syncthreads();
  if (t < 8) {
    float s = b1[t];
    for (int j = 0; j < 128; ++j) s += w1[t * 128 + j] * gv[j];
    hid[t] = (s >= 0.f) ? s : 0.1f * s;
  }
  __syncthreads();
  float s = b2[t];
  for (int j = 0; j < 8; ++j) s += w2[t * 8 + j] * hid[j];
  sc[t] = 1.f / (1.f + __expf(-s));
}

__global__ void k_scale_bf(unsigned short* __restrict__ x,
                           const float* __restrict__ sc, int C, int total) {
  int i = blockIdx.x * blockDim.x + threadIdx.x;
  if (i < total) x[i] = f2bf(bf2f(x[i]) * sc[i % C]);
}

__global__ void k_pshuf_bf(const unsigned short* __restrict__ in,
                           unsigned short* __restrict__ out, int H, int W) {
  int H2 = H * 2, W2 = W * 2;
  int total = 64 * H2 * W2;
  int i = blockIdx.x * blockDim.x + threadIdx.x;
  if (i >= total) return;
  int c = i & 63;
  int r = i >> 6;
  int ow = r % W2, oh = r / W2;
  out[i] = in[((size_t)(oh >> 1) * W + (ow >> 1)) * 256 + c * 4 + (oh & 1) * 2 + (ow & 1)];
}

__device__ inline float cubw(float s) {
  s = fabsf(s);
  const float a = -0.5f;
  if (s <= 1.f) return ((a + 2.f) * s - (a + 3.f)) * s * s + 1.f;
  if (s < 2.f) return (((s - 5.f) * s + 8.f) * s - 4.f) * a;
  return 0.f;
}

__global__ void k_bicubic2x(const float* __restrict__ x, float* __restrict__ y,
                            int H, int W) {
  int H2 = H * 2, W2 = W * 2;
  int total = 3 * H2 * W2;
  int i = blockIdx.x * blockDim.x + threadIdx.x;
  if (i >= total) return;
  int c = i / (H2 * W2);
  int r = i - c * (H2 * W2);
  int oh = r / W2, ow = r - (r / W2) * W2;
  float iy = (oh + 0.5f) * 0.5f - 0.5f;
  float ix = (ow + 0.5f) * 0.5f - 0.5f;
  int y0 = (int)floorf(iy), x0 = (int)floorf(ix);
  float fy = iy - y0, fx = ix - x0;
  float acc = 0.f;
  const float* xc = x + (size_t)c * H * W;
#pragma unroll
  for (int dy = 0; dy < 4; ++dy) {
    int yy = y0 - 1 + dy;
    yy = yy < 0 ? 0 : (yy > H - 1 ? H - 1 : yy);
    float wy = cubw(fy - (float)(dy - 1));
#pragma unroll
    for (int dx = 0; dx < 4; ++dx) {
      int xx = x0 - 1 + dx;
      xx = xx < 0 ? 0 : (xx > W - 1 ? W - 1 : xx);
      acc += wy * cubw(fx - (float)(dx - 1)) * xc[(size_t)yy * W + xx];
    }
  }
  y[i] = acc;
}

// weight permute+pad: OIHW fp32 -> [co][(kh*ks+kw)*Cpad + ci] bf16, zero-padded
__global__ void k_wperm(const float* __restrict__ src,
                        unsigned short* __restrict__ dst, int CoutT, int Cin,
                        int Cpad, int ks) {
  int ks2 = ks * ks;
  int per = ks2 * Cpad;
  int i = blockIdx.x * blockDim.x + threadIdx.x;
  if (i >= CoutT * per) return;
  int co = i / per;
  int rem = i - co * per;
  int q = rem / Cpad;
  int ci = rem - q * Cpad;
  int kh = q / ks, kw = q - kh * ks;
  float v = 0.f;
  if (ci < Cin) v = src[(((size_t)co * Cin + ci) * ks + kh) * ks + kw];
  dst[i] = f2bf(v);
}

// ------------------------- WMMA implicit-GEMM conv -------------------------
// NHWC bf16. LDS: raw input rows [kh][PXT][Cpad] (async DMA) + a 64xCpad
// weight tap-tile (TDM DMA, one per (kh,kw)). Block = 8 wave32 -> 64x32 tile.
// flags: 1=lrelu, 2=residual, 4=fp32 NCHW output (final layer, resF/outF).
__global__ void __launch_bounds__(256)
k_conv_wmma(const unsigned short* __restrict__ in, int inS,
            const unsigned short* __restrict__ wbf, const float* __restrict__ bias,
            const unsigned short* __restrict__ res, int resS,
            unsigned short* __restrict__ out, int outS,
            const float* __restrict__ resF, float* __restrict__ outF,
            int H, int W, int Cin, int Cpad, int Cout, int ks, int pad,
            int outChOff, int flags) {
  extern __shared__ unsigned short sIn[];  // [ks][PXT][Cpad] + [64][Cpad]
  const int HW = H * W;
  const int g = blockIdx.z;
  in += g * Cin;
  wbf += (size_t)g * Cout * (ks * ks * Cpad);
  bias += (size_t)g * Cout;
  const int outCh0 = outChOff + g * Cout;
  const int resCh0 = g * Cout;

  const int PXT = 32 + 2 * pad;
  const int tpr = W >> 5;
  const int px0 = (blockIdx.x % tpr) << 5;
  const int h = blockIdx.x / tpr;
  const int co0 = blockIdx.y << 6;

  const int wave = threadIdx.x >> 5, lane = threadIdx.x & 31;
  const int wm = wave & 3, wn = wave >> 2;
  const int rowBase = co0 + (wm << 4);
  const int nBase = wn << 4;
  const int l15 = lane & 15, lh = lane >> 4;

  // ---- stage raw input rows via async global->LDS DMA; zero-fill OOB ----
  const int CPP = Cpad >> 2;
  const int total = ks * PXT * CPP;
  for (int t = threadIdx.x; t < total; t += 256) {
    int kh = t / (PXT * CPP);
    int rem = t - kh * (PXT * CPP);
    int px = rem / CPP;
    int qw = rem - px * CPP;
    int ci0 = qw << 2;
    unsigned short* lptr = sIn + ((size_t)(kh * PXT + px) * Cpad + ci0);
    int ih = h + kh - pad, iw = px0 + px - pad;
    if ((unsigned)ih < (unsigned)H && (unsigned)iw < (unsigned)W && ci0 < Cin) {
      async_b64(in + (size_t)(ih * W + iw) * inS + ci0, lptr);
    } else {
      ((unsigned*)lptr)[0] = 0u;
      ((unsigned*)lptr)[1] = 0u;
    }
  }
  wait_async0();
  __syncthreads();

  const int KtotP = ks * ks * Cpad;
#if HAVE_TDM
  unsigned short* sW = sIn + (size_t)ks * PXT * Cpad;   // [64][Cpad] weight tile
  const unsigned short* arowL = sW + (size_t)(wm * 16 + l15) * Cpad;
#else
  const unsigned short* arowG = wbf + (size_t)(rowBase + l15) * KtotP;
  const bool rok = (rowBase + l15) < Cout;
  __builtin_prefetch(arowG, 0, 1);
#endif

  v8f acc = vzero8();
  for (int kh = 0; kh < ks; ++kh) {
    for (int kw = 0; kw < ks; ++kw) {
      const int q = kh * ks + kw;
#if HAVE_TDM
      // wave0 DMAs the 64-row tap tile; OOB rows (>=Cout-co0) zero-filled
      if (threadIdx.x < 32) {
        tdm_load_2d(wbf + (size_t)co0 * KtotP + (size_t)q * Cpad, sW, Cpad, 64,
                    Cout - co0, KtotP);
        wait_tensor0();
      }
      __syncthreads();
#endif
      const unsigned short* brow =
          sIn + (size_t)(kh * PXT + nBase + l15 + kw) * Cpad;
      for (int cb = 0; cb < Cpad; cb += 32) {
#if HAVE_TDM
        bf16x16 av = fragA_mem(arowL, cb, lane, true);
#else
        bf16x16 av = fragA_mem(arowG + q * Cpad, cb, lane, rok);
#endif
        bf16x16 bv = fragB_mem(brow, cb, lane);
        acc = wmma_bf16(av, bv, acc);
      }
#if HAVE_TDM
      __syncthreads();                         // WAR: before next TDM overwrite
#endif
    }
  }

  // ---- epilogue ----
  const int n = nBase + l15;
  const int w = px0 + n;
  const size_t pix = (size_t)h * W + w;
  if (flags & 4) {                              // fp32 NCHW output path
#pragma unroll
    for (int r = 0; r < 8; ++r) {
      int co = rowBase + r + (lh << 3);
      if (co < Cout && w < W) {
        float v = acc[r] + bias[co];
        if (flags & 1) v = (v >= 0.f) ? v : 0.1f * v;
        if (flags & 2) v += resF[(size_t)(resCh0 + co) * HW + pix];
        outF[(size_t)(outCh0 + co) * HW + pix] = v;
      }
    }
  } else {                                      // bf16 NHWC, packed pairs
    const size_t pbase = pix * (size_t)outS + outCh0;
#pragma unroll
    for (int r = 0; r < 8; r += 2) {
      int co = rowBase + r + (lh << 3);
      if (co < Cout && w < W) {
        float v0 = acc[r] + bias[co];
        float v1 = acc[r + 1] + bias[co + 1];
        if (flags & 1) {
          v0 = (v0 >= 0.f) ? v0 : 0.1f * v0;
          v1 = (v1 >= 0.f) ? v1 : 0.1f * v1;
        }
        if (flags & 2) {
          v0 += bf2f(res[pix * (size_t)resS + resCh0 + co]);
          v1 += bf2f(res[pix * (size_t)resS + resCh0 + co + 1]);
        }
        unsigned pk = (unsigned)f2bf(v0) | ((unsigned)f2bf(v1) << 16);
        *(unsigned*)(out + pbase + co) = pk;
      }
    }
  }
}

// ----------------------- PAM window attention (WMMA) -----------------------
__global__ void __launch_bounds__(256)
k_pam_attn(const unsigned short* __restrict__ fa, const unsigned short* __restrict__ fb,
           const unsigned short* __restrict__ xg, const unsigned short* __restrict__ xb,
           const float* __restrict__ disp, unsigned short* __restrict__ out,
           int H, int W, int dir) {
  __shared__ unsigned short sA[64 * 64];
  __shared__ unsigned short sBm[64 * 64];
  __shared__ unsigned short sP[64 * 64];
  __shared__ unsigned short sX[64 * 64];
  __shared__ float sS[64 * 64];
  __shared__ float sMean[64];
  __shared__ float sMask[64];
  __shared__ int sIdx[64];

  const int wpr = W >> 3;
  const int w0 = (blockIdx.x % wpr) << 3;
  const int h0 = (blockIdx.x / wpr) << 3;
  const int tid = threadIdx.x;

  if (tid < 64) {
    int py = tid >> 3, px = tid & 7;
    int w = w0 + px;
    int d = (int)disp[(size_t)(h0 + py) * W + w];
    int t = dir ? (w + d) : (w - d);
    int idx = t < 0 ? 0 : (t > W - 1 ? W - 1 : t);
    sIdx[tid] = idx;
    sMask[tid] = (dir ? (t <= W - 1) : (t >= 0)) ? 1.f : 0.f;
  }
  __syncthreads();

  for (int i = tid; i < 4096; i += 256) {
    int p = i >> 6, c = i & 63;
    int py = p >> 3, px = p & 7;
    size_t rowb = ((size_t)(h0 + py) * W);
    sA[i] = fa[(rowb + w0 + px) * 64 + c];
    sS[i] = bf2f(fb[(rowb + sIdx[p]) * 64 + c]);
    sX[c * 64 + p] = xg[(rowb + sIdx[p]) * 64 + c];
  }
  __syncthreads();

  if (tid < 64) {
    float s = 0.f;
    for (int p = 0; p < 64; ++p) s += sS[p * 64 + tid];
    sMean[tid] = s * (1.f / 64.f);
  }
  __syncthreads();
  for (int i = tid; i < 4096; i += 256) sBm[i] = f2bf(sS[i] - sMean[i & 63]);
  __syncthreads();

  const int lane = tid & 31, wave = tid >> 5;
  const int l15 = lane & 15, lh = lane >> 4;

#pragma unroll
  for (int t2 = 0; t2 < 2; ++t2) {               // scores
    int t = wave * 2 + t2;
    int tm = (t >> 2) << 4, tn = (t & 3) << 4;
    v8f acc = vzero8();
#pragma unroll
    for (int kb = 0; kb < 64; kb += 32)
      acc = wmma_bf16(fragA_mem(sA + (size_t)(tm + l15) * 64, kb, lane, true),
                      fragB_mem(sBm + (size_t)(tn + l15) * 64, kb, lane), acc);
#pragma unroll
    for (int r = 0; r < 8; ++r)
      sS[(tm + r + (lh << 3)) * 64 + tn + l15] = acc[r];
  }
  __syncthreads();

  if (tid < 64) {                                 // row softmax
    float mx = -1e30f;
    for (int j = 0; j < 64; ++j) mx = fmaxf(mx, sS[tid * 64 + j]);
    float sum = 0.f;
    for (int j = 0; j < 64; ++j) {
      float e = __expf(sS[tid * 64 + j] - mx);
      sS[tid * 64 + j] = e;
      sum += e;
    }
    float inv = 1.f / sum;
    for (int j = 0; j < 64; ++j) sP[tid * 64 + j] = f2bf(sS[tid * 64 + j] * inv);
  }
  __syncthreads();

#pragma unroll
  for (int t2 = 0; t2 < 2; ++t2) {               // O = P @ Xg
    int t = wave * 2 + t2;
    int tm = (t >> 2) << 4, tn = (t & 3) << 4;
    v8f acc = vzero8();
#pragma unroll
    for (int kb = 0; kb < 64; kb += 32)
      acc = wmma_bf16(fragA_mem(sP + (size_t)(tm + l15) * 64, kb, lane, true),
                      fragB_mem(sX + (size_t)(tn + l15) * 64, kb, lane), acc);
#pragma unroll
    for (int r = 0; r < 8; ++r) {
      int m = tm + r + (lh << 3);
      int nn = tn + l15;
      int py = m >> 3, px = m & 7;
      size_t o = ((size_t)(h0 + py) * W + (w0 + px)) * 64 + nn;
      out[o] = f2bf(bf2f(xb[o]) + acc[r] * sMask[m]);
    }
  }
}

// --------------------------------- launch ---------------------------------

extern "C" void kernel_launch(void* const* d_in, const int* in_sizes, int n_in,
                              void* d_out, int out_size, void* d_ws, size_t ws_size,
                              hipStream_t stream) {
  (void)in_sizes; (void)n_in; (void)out_size; (void)ws_size;
  const int H = 128, W = 384, HW = H * W;
  const int H2 = 256, W2 = 768, HW2 = H2 * W2;

  const float* x_left  = (const float*)d_in[0];
  const float* x_right = (const float*)d_in[1];

  size_t wsoff = 0;
  auto alloc = [&](size_t bytes) -> void* {
    size_t a = (wsoff + 255) & ~(size_t)255;
    wsoff = a + bytes;
    return (void*)((char*)d_ws + a);
  };
  auto hbuf = [&](size_t n) { return (unsigned short*)alloc(n * 2); };
  auto fbuf = [&](size_t n) { return (float*)alloc(n * sizeof(float)); };

  // ---- params (depth-first, dict-insertion order) ----
  int ci = 2;
  auto rw = [&]() { return (const float*)d_in[ci++]; };
  struct RC { const float* w; const float* b; int cout, cin, ks; };
  auto rconv = [&](int cout, int cin, int ks) {
    RC r; r.w = rw(); r.b = rw(); r.cout = cout; r.cin = cin; r.ks = ks; return r;
  };
  struct CW { const unsigned short* w; const float* b; int cing, cpad, coutg, ks; };
  auto mk = [&](RC r, int groups) -> CW {
    int cpad = ((r.cin + 31) / 32) * 32;
    size_t e = (size_t)r.cout * r.ks * r.ks * cpad;
    unsigned short* dw = (unsigned short*)alloc(e * 2);
    k_wperm<<<(int)((e + 255) / 256), 256, 0, stream>>>(r.w, dw, r.cout, r.cin,
                                                        cpad, r.ks);
    CW c; c.w = dw; c.b = r.b; c.cing = r.cin; c.cpad = cpad;
    c.coutg = r.cout / groups; c.ks = r.ks;
    return c;
  };

  CW initw = mk(rconv(64, 4, 3), 1);
  CW deepw[4][5];
  for (int rj = 0; rj < 4; ++rj) {
    for (int i = 0; i < 4; ++i) deepw[rj][i] = mk(rconv(24, 64 + 24 * i, 3), 1);
    deepw[rj][4] = mk(rconv(64, 160, 1), 1);
  }
  CW deepc = mk(rconv(64, 256, 1), 1);
  const float* bn_g = rw();
  const float* bn_b = rw();
  CW rb1 = mk(rconv(256, 64, 3), 4);
  CW rb2 = mk(rconv(256, 64, 3), 4);
  CW bq  = mk(rconv(64, 64, 1), 4);
  CW bs  = mk(rconv(64, 64, 1), 4);
  CW fusw[5];
  for (int i = 0; i < 4; ++i) fusw[i] = mk(rconv(32, 128 + 32 * i, 3), 1);
  fusw[4] = mk(rconv(128, 256, 1), 1);
  RC ca1 = rconv(8, 128, 1);
  RC ca2 = rconv(128, 8, 1);
  CW fusc = mk(rconv(64, 128, 1), 1);
  CW recw[4][5];
  for (int rj = 0; rj < 4; ++rj) {
    for (int i = 0; i < 4; ++i) recw[rj][i] = mk(rconv(24, 64 + 24 * i, 3), 1);
    recw[rj][4] = mk(rconv(64, 160, 1), 1);
  }
  CW recc = mk(rconv(64, 256, 1), 1);
  CW up1w = mk(rconv(256, 64, 1), 1);
  CW up2w = mk(rconv(3, 64, 3), 1);

  // ---- activation buffers (bf16 NHWC) ----
  unsigned short* xlh  = hbuf((size_t)4 * HW);
  unsigned short* xrh  = hbuf((size_t)4 * HW);
  unsigned short* bL   = hbuf((size_t)64 * HW);
  unsigned short* bR   = hbuf((size_t)64 * HW);
  unsigned short* catL = hbuf((size_t)256 * HW);
  unsigned short* catR = hbuf((size_t)256 * HW);
  unsigned short* work = hbuf((size_t)256 * HW);
  unsigned short* tb   = hbuf((size_t)256 * HW);
  unsigned short* tc   = hbuf((size_t)256 * HW);
  unsigned short* Qf   = hbuf((size_t)64 * HW);
  unsigned short* Kf   = hbuf((size_t)64 * HW);
  unsigned short* pl   = hbuf((size_t)64 * HW);
  unsigned short* pr   = hbuf((size_t)64 * HW);
  unsigned short* z    = hbuf((size_t)128 * HW);
  unsigned short* f64  = hbuf((size_t)64 * HW);
  unsigned short* recb = hbuf((size_t)64 * HW);
  unsigned short* ub64 = hbuf((size_t)64 * HW2);
  float* base = fbuf((size_t)3 * HW2);
  float* gvec = fbuf(128);
  float* scv  = fbuf(128);

  auto conv = [&](const unsigned short* in, int inS, const CW& c,
                  const unsigned short* res, int resS, unsigned short* out,
                  int outS, int chOff, int flags, int groups, int Hh, int Ww,
                  const float* resF = nullptr, float* outF = nullptr) {
    int pad = c.ks / 2;
    int PXT = 32 + 2 * pad;
    dim3 grid((Ww / 32) * Hh, (c.coutg + 63) / 64, groups);
    size_t sm = (size_t)c.ks * PXT * c.cpad * 2;
#if 1  // weight tap-tile staged in LDS by the TDM (when builtin available)
    sm += (size_t)64 * c.cpad * 2;
#endif
    k_conv_wmma<<<grid, 256, sm, stream>>>(in, inS, c.w, c.b, res, resS, out, outS,
                                           resF, outF, Hh, Ww, c.cing, c.cpad,
                                           c.coutg, c.ks, pad, chOff, flags);
  };
  auto catcpy = [&](const unsigned short* s, int sS, unsigned short* d, int dS,
                    int dOff, int C) {
    int n = HW * C;
    k_catcpy<<<(n + 255) / 256, 256, 0, stream>>>(s, sS, d, dS, dOff, C, HW);
  };

  auto run_rdg = [&](const unsigned short* x, int xS, CW (&wv)[4][5], const CW& gc,
                     unsigned short* cat, unsigned short* outb) {
    const unsigned short* cur = x;
    int curS = xS;
    for (int rj = 0; rj < 4; ++rj) {
      catcpy(cur, curS, work, 256, 0, 64);
      int cch = 64;
      for (int i = 0; i < 4; ++i) {
        conv(work, 256, wv[rj][i], nullptr, 0, work, 256, cch, 1, 1, H, W);
        cch += 24;
      }
      conv(work, 256, wv[rj][4], cur, curS, cat, 256, rj * 64, 2, 1, H, W);
      cur = cat + rj * 64;
      curS = 256;
    }
    conv(cat, 256, gc, nullptr, 0, outb, 64, 0, 0, 1, H, W);
  };

  auto feat = [&](const unsigned short* cat, const CW& head, unsigned short* outf) {
    int tot = 256 * HW;
    k_affine_bf<<<(tot + 255) / 256, 256, 0, stream>>>(cat, bn_g, bn_b, work, 256, tot);
    conv(work, 256, rb1, nullptr, 0, tb, 256, 0, 1, 4, H, W);
    conv(tb, 256, rb2, work, 256, tc, 256, 0, 2, 4, H, W);
    conv(tc, 256, head, nullptr, 0, outf, 64, 0, 0, 4, H, W);
  };

  auto fuse = [&](const unsigned short* b64, const unsigned short* pT,
                  unsigned short* out64) {
    catcpy(b64, 64, work, 256, 0, 64);
    catcpy(pT, 64, work, 256, 64, 64);
    int cch = 128;
    for (int i = 0; i < 4; ++i) {
      conv(work, 256, fusw[i], nullptr, 0, work, 256, cch, 1, 1, H, W);
      cch += 32;
    }
    conv(work, 256, fusw[4], work, 256, z, 128, 0, 2, 1, H, W);
    k_gap_bf<<<128, 256, 0, stream>>>(z, gvec, 128, HW);
    k_ca_mlp<<<1, 128, 0, stream>>>(gvec, ca1.w, ca1.b, ca2.w, ca2.b, scv);
    k_scale_bf<<<(128 * HW + 255) / 256, 256, 0, stream>>>(z, scv, 128, 128 * HW);
    conv(z, 128, fusc, nullptr, 0, out64, 64, 0, 0, 1, H, W);
  };

  auto tail = [&](const float* xin, const unsigned short* bbuf,
                  const unsigned short* pT, unsigned short* cat, float* outp) {
    k_bicubic2x<<<(3 * HW2 + 255) / 256, 256, 0, stream>>>(xin, base, H, W);
    fuse(bbuf, pT, f64);
    run_rdg(f64, 64, recw, recc, cat, recb);
    conv(recb, 64, up1w, nullptr, 0, work, 256, 0, 0, 1, H, W);
    k_pshuf_bf<<<(64 * HW2 + 255) / 256, 256, 0, stream>>>(work, ub64, H, W);
    conv(ub64, 64, up2w, nullptr, 0, nullptr, 0, 0, 2 | 4, 1, H2, W2, base, outp);
  };

  // ---------------------------- network schedule ---------------------------
  k_nchw2nhwc<<<(4 * HW + 255) / 256, 256, 0, stream>>>(x_left, xlh, 4, HW);
  k_nchw2nhwc<<<(4 * HW + 255) / 256, 256, 0, stream>>>(x_right, xrh, 4, HW);
  conv(xlh, 4, initw, nullptr, 0, bL, 64, 0, 0, 1, H, W);
  conv(xrh, 4, initw, nullptr, 0, bR, 64, 0, 0, 1, H, W);
  run_rdg(bL, 64, deepw, deepc, catL, bL);
  run_rdg(bR, 64, deepw, deepc, catR, bR);

  feat(catL, bq, Qf);
  feat(catR, bs, Kf);
  k_pam_attn<<<(W / 8) * (H / 8), 256, 0, stream>>>(
      Qf, Kf, bR, bL, x_left + (size_t)3 * HW, pl, H, W, 0);
  k_pam_attn<<<(W / 8) * (H / 8), 256, 0, stream>>>(
      Kf, Qf, bL, bR, x_right + (size_t)3 * HW, pr, H, W, 1);

  float* out = (float*)d_out;
  tail(x_left,  bL, pl, catL, out);
  tail(x_right, bR, pr, catR, out + (size_t)3 * HW2);
}
